// KnowledgeEmbeddings_69458211111435
// MI455X (gfx1250) — matrix operand order repaired and checked by reading, not compiled, not run on hardware
//
#include <hip/hip_runtime.h>
#include <hip/hip_bf16.h>

// ---------------------------------------------------------------------------
// KnowledgeEmbeddings for MI455X (gfx1250, wave32, WMMA).
// Dominant cost: 137 GF logits GEMM + 2x17 GF projections -> bf16 WMMA.
// M = B*L = 8192, D = 1024, NV = 8192, TOP_K = 32.
// ---------------------------------------------------------------------------

#define MM   8192   // B*L token rows
#define DD   1024   // model dim
#define NVV  8192   // knowledge rows
#define TOPK 32

typedef __attribute__((ext_vector_type(16))) __bf16 v16bf;
typedef __attribute__((ext_vector_type(8)))  float  v8f;

union FragBF { v16bf v; unsigned u[8]; };

__device__ __forceinline__ unsigned short f32_to_bf16_rne(float x) {
    unsigned u = __float_as_uint(x);
    u += 0x7fffu + ((u >> 16) & 1u);   // round-to-nearest-even
    return (unsigned short)(u >> 16);
}

// ---------------------------------------------------------------------------
// fp32 -> bf16 bulk convert (grid-stride)
// ---------------------------------------------------------------------------
__global__ __launch_bounds__(256) void cvt_f32_bf16(const float* __restrict__ src,
                                                    unsigned short* __restrict__ dst,
                                                    int n) {
    int i = blockIdx.x * 256 + threadIdx.x;
    int stride = gridDim.x * 256;
    for (; i < n; i += stride) dst[i] = f32_to_bf16_rne(src[i]);
}

// ---------------------------------------------------------------------------
// bf16 WMMA GEMM:  C[M,N] = A[M,K] (rm) * B[K,N] (rm) + bias
// Block tile 64x64, 8 waves x two 16x16 accumulators, K swept 32/step via LDS.
// B is staged TRANSPOSED + k-pair packed so both fragment loads are plain
// consecutive-dword ds_load_b128 pairs.
//   As[row][kp] : (A[row][2kp],  A[row][2kp+1])
//   Bs[col][kp] : (B[2kp][col],  B[2kp+1][col])
// ---------------------------------------------------------------------------
template <bool F32OUT>
__global__ __launch_bounds__(256) void gemm_bf16_wmma(
    const unsigned short* __restrict__ A,
    const unsigned short* __restrict__ B,
    const float* __restrict__ bias,
    float* __restrict__ outF,
    unsigned short* __restrict__ outH,
    int M, int N, int K)
{
    __shared__ unsigned As[64 * 16];     // [row][kpair]
    __shared__ unsigned Bs[64 * 16];     // [col][kpair]  (transposed, packed)

    const int tid  = threadIdx.x;
    const int lane = tid & 31;
    const int wave = tid >> 5;
    const int m0   = blockIdx.y * 64;
    const int n0   = blockIdx.x * 64;
    const int sr   = wave & 3;           // subtile row   0..3
    const int sc0  = (wave >> 2) * 2;    // subtile col base {0,2}
    const int ml   = lane & 15;
    const int hl   = lane >> 4;

    const float b0v = bias[n0 + sc0 * 16 + ml];
    const float b1v = bias[n0 + (sc0 + 1) * 16 + ml];
    v8f c0, c1;
#pragma unroll
    for (int v = 0; v < 8; ++v) { c0[v] = b0v; c1[v] = b1v; }

    for (int kt = 0; kt < K; kt += 32) {
        // ---- stage A 64x32: 256 uint4, b128 loads, 1/thread ----
        {
            int r = tid >> 2, q4 = tid & 3;
            ((uint4*)As)[tid] =
                *(const uint4*)(A + (size_t)(m0 + r) * K + kt + q4 * 8);
        }
        // ---- stage B 32x64 -> transposed packed: 512 row-pair ids, 2/thread
#pragma unroll
        for (int i = 0; i < 2; ++i) {
            int id = tid + i * 256;
            int np = id & 31;            // column pair (coalesced across lanes)
            int kp = id >> 5;            // k row-pair
            const unsigned short* gb =
                B + (size_t)(kt + 2 * kp) * N + n0 + 2 * np;
            unsigned a = *(const unsigned*)(gb);       // (B[k][n], B[k][n+1])
            unsigned b = *(const unsigned*)(gb + N);   // (B[k+1][n], B[k+1][n+1])
            Bs[(2 * np)     * 16 + kp] = (a & 0xffffu) | (b << 16);
            Bs[(2 * np + 1) * 16 + kp] = (a >> 16)     | (b & 0xffff0000u);
        }
        __syncthreads();

        FragBF a, fb0, fb1;
#pragma unroll
        for (int v = 0; v < 4; ++v) {
            a.u[v]     = As[(sr * 16 + ml) * 16 + hl * 8 + v];
            a.u[v + 4] = As[(sr * 16 + ml) * 16 + hl * 8 + 4 + v];
        }
        const int nc0 = sc0 * 16 + ml;
        const int nc1 = nc0 + 16;
#pragma unroll
        for (int v = 0; v < 8; ++v) {
            fb0.u[v] = Bs[nc0 * 16 + hl * 8 + v];
            fb1.u[v] = Bs[nc1 * 16 + hl * 8 + v];
        }
        c0 = __builtin_amdgcn_wmma_f32_16x16x32_bf16(false, a.v, false, fb0.v,
                                                     (short)0, c0, false, false);
        c1 = __builtin_amdgcn_wmma_f32_16x16x32_bf16(false, a.v, false, fb1.v,
                                                     (short)0, c1, false, false);
        __syncthreads();
    }

    // C/D layout: vgpr v -> row v + hl*8 ; col = lane&15
#pragma unroll
    for (int v = 0; v < 8; ++v) {
        size_t row = (size_t)(m0 + sr * 16 + v + hl * 8);
        int col0 = n0 + sc0 * 16 + ml;
        int col1 = col0 + 16;
        if constexpr (F32OUT) {
            outF[row * N + col0] = c0[v];
            outF[row * N + col1] = c1[v];
        } else {
            outH[row * N + col0] = f32_to_bf16_rne(c0[v]);
            outH[row * N + col1] = f32_to_bf16_rne(c1[v]);
        }
    }
}

// ---------------------------------------------------------------------------
// Fused attention logits + top-32 + softmax.
// One block per 32 q rows (halves knowledge-bank re-reads vs 16:
// 256 blocks x 16 MB = 4 GB of L2 traffic).  32x1024 bf16 A panel staged once
// in LDS (64 KB of the 320 KB WGP).  NV swept in 256-col chunks; a bf16
// knowledge row is already in B-fragment (k,k+1)-packed order, so B frags load
// straight from global as b128.  2x2 accumulator grid per wave: every A and B
// fragment feeds two WMMAs (4 wmma per 4 global_b128 + 4 ds_b128).
// Top-k: wave w owns rows 4w..4w+3; lanes test 32 logits at a time against the
// broadcast threshold, __ballot (wave32) compresses survivors, lane 0 inserts.
// ---------------------------------------------------------------------------
__global__ __launch_bounds__(256) void attn_topk_softmax(
    const unsigned short* __restrict__ Q,    // [M,D]  bf16
    const unsigned short* __restrict__ Kb,   // [NV,D] bf16
    float* __restrict__ topw,                // [M,TOPK]
    int*   __restrict__ topi)                // [M,TOPK]
{
    __shared__ unsigned Qs[32 * 512];        // 32 rows x 1024 bf16 = 64 KB
    __shared__ float    Ls[32 * 256];        // logits chunk, 32 KB
    __shared__ float    tv[32 * TOPK];       // 4 KB
    __shared__ int      ti[32 * TOPK];       // 4 KB

    const int tid  = threadIdx.x;
    const int lane = tid & 31;
    const int wave = tid >> 5;
    const int m0   = blockIdx.x * 32;
    const int ml   = lane & 15;
    const int hl   = lane >> 4;
    const int sub0 = wave * 2;
    const int sub1 = wave * 2 + 1;

    // stage Q panel: 4096 uint4 (b128), 16/thread, coalesced
    for (int i = tid; i < 32 * 128; i += 256) {
        int row = i >> 7;                // 128 uint4 per 1024-bf16 row
        int c4  = i & 127;
        ((uint4*)Qs)[i] =
            *(const uint4*)(Q + (size_t)(m0 + row) * DD + c4 * 8);
    }
    if (tid < 32) {
#pragma unroll
        for (int j = 0; j < TOPK; ++j) { tv[tid * TOPK + j] = -3.4e38f; ti[tid * TOPK + j] = 0; }
    }
    __syncthreads();

    float cm[4] = {-3.4e38f, -3.4e38f, -3.4e38f, -3.4e38f};
    int   cp[4] = {0, 0, 0, 0};

    for (int chunk = 0; chunk < NVV; chunk += 256) {
        // prefetch next chunk's knowledge rows (global_prefetch_b8)
        if (chunk + 256 < NVV)
            __builtin_prefetch(Kb + (size_t)(chunk + 256 + tid) * DD, 0, 3);

        v8f c00, c01, c10, c11;
#pragma unroll
        for (int v = 0; v < 8; ++v) { c00[v] = 0.f; c01[v] = 0.f; c10[v] = 0.f; c11[v] = 0.f; }
        const size_t n0g = (size_t)(chunk + sub0 * 16 + ml);
        const size_t n1g = (size_t)(chunk + sub1 * 16 + ml);

#pragma unroll 4
        for (int kk = 0; kk < DD; kk += 32) {
            FragBF a0, a1, fb0, fb1;
#pragma unroll
            for (int v = 0; v < 4; ++v) {
                a0.u[v]     = Qs[ml * 512 + (kk >> 1) + hl * 8 + v];
                a0.u[v + 4] = Qs[ml * 512 + (kk >> 1) + hl * 8 + 4 + v];
                a1.u[v]     = Qs[(16 + ml) * 512 + (kk >> 1) + hl * 8 + v];
                a1.u[v + 4] = Qs[(16 + ml) * 512 + (kk >> 1) + hl * 8 + 4 + v];
            }
#pragma unroll
            for (int v = 0; v < 8; ++v) {
                int k = kk + hl * 16 + 2 * v;   // consecutive dwords -> b128
                fb0.u[v] = *(const unsigned*)(Kb + n0g * DD + k);
                fb1.u[v] = *(const unsigned*)(Kb + n1g * DD + k);
            }
            c00 = __builtin_amdgcn_wmma_f32_16x16x32_bf16(false, a0.v, false, fb0.v,
                                                          (short)0, c00, false, false);
            c01 = __builtin_amdgcn_wmma_f32_16x16x32_bf16(false, a0.v, false, fb1.v,
                                                          (short)0, c01, false, false);
            c10 = __builtin_amdgcn_wmma_f32_16x16x32_bf16(false, a1.v, false, fb0.v,
                                                          (short)0, c10, false, false);
            c11 = __builtin_amdgcn_wmma_f32_16x16x32_bf16(false, a1.v, false, fb1.v,
                                                          (short)0, c11, false, false);
        }
        // scatter logits to LDS, scaled by 1/sqrt(D) = 1/32
#pragma unroll
        for (int v = 0; v < 8; ++v) {
            int row = v + hl * 8;
            Ls[row * 256 + sub0 * 16 + ml]        = c00[v] * 0.03125f;
            Ls[row * 256 + sub1 * 16 + ml]        = c01[v] * 0.03125f;
            Ls[(16 + row) * 256 + sub0 * 16 + ml] = c10[v] * 0.03125f;
            Ls[(16 + row) * 256 + sub1 * 16 + ml] = c11[v] * 0.03125f;
        }
        __syncthreads();

        // ---- wave-parallel top-32 maintenance: wave owns 4 rows ----
#pragma unroll
        for (int rr = 0; rr < 4; ++rr) {
            const int row = wave * 4 + rr;
            float* rv = &tv[row * TOPK];
            int*   ri = &ti[row * TOPK];
            const float* lr = &Ls[row * 256];
#pragma unroll
            for (int i = 0; i < 8; ++i) {
                float x = lr[i * 32 + lane];
                unsigned m = (unsigned)__ballot(x > cm[rr]);
                if (lane == 0) {
                    while (m) {
                        int b = __builtin_ctz(m);
                        m &= m - 1;
                        float xx = lr[i * 32 + b];
                        if (xx > cm[rr]) {
                            rv[cp[rr]] = xx;
                            ri[cp[rr]] = chunk + i * 32 + b;
                            cm[rr] = rv[0]; cp[rr] = 0;
#pragma unroll
                            for (int t = 1; t < TOPK; ++t)
                                if (rv[t] < cm[rr]) { cm[rr] = rv[t]; cp[rr] = t; }
                        }
                    }
                }
                cm[rr] = __shfl(cm[rr], 0, 32);
            }
        }
        __syncthreads();
    }

    // softmax over the 32 survivors
    if (tid < 32) {
        float* rv = &tv[tid * TOPK];
        int*   ri = &ti[tid * TOPK];
        float mx = rv[0];
#pragma unroll
        for (int t = 1; t < TOPK; ++t) mx = fmaxf(mx, rv[t]);
        float e[TOPK], s = 0.0f;
#pragma unroll
        for (int t = 0; t < TOPK; ++t) { e[t] = __expf(rv[t] - mx); s += e[t]; }
        float inv = 1.0f / s;
        size_t base = (size_t)(m0 + tid) * TOPK;
#pragma unroll
        for (int t = 0; t < TOPK; ++t) { topw[base + t] = e[t] * inv; topi[base + t] = ri[t]; }
    }
}

// ---------------------------------------------------------------------------
// retrieved[row] = sum_k w_k * knowledge[idx_k]   (fp32 accuracy),
// plus bf16 copy for the Wo WMMA GEMM.  One block per row.
// ---------------------------------------------------------------------------
__global__ __launch_bounds__(256) void gather_weighted(
    const float* __restrict__ knowledge,
    const float* __restrict__ topw, const int* __restrict__ topi,
    float* __restrict__ retF, unsigned short* __restrict__ retH)
{
    __shared__ float sw[TOPK];
    __shared__ int   si[TOPK];
    const int row = blockIdx.x;
    const int tid = threadIdx.x;
    if (tid < TOPK) {
        sw[tid] = topw[(size_t)row * TOPK + tid];
        si[tid] = topi[(size_t)row * TOPK + tid];
    }
    __syncthreads();
    float acc[4] = {0.f, 0.f, 0.f, 0.f};
    for (int k = 0; k < TOPK; ++k) {
        float w = sw[k];
        const float* kr = knowledge + (size_t)si[k] * DD;
#pragma unroll
        for (int c = 0; c < 4; ++c) acc[c] += w * kr[tid + c * 256];
    }
#pragma unroll
    for (int c = 0; c < 4; ++c) {
        retF[(size_t)row * DD + tid + c * 256] = acc[c];
        retH[(size_t)row * DD + tid + c * 256] = f32_to_bf16_rne(acc[c]);
    }
}

// ---------------------------------------------------------------------------
// gate = sigmoid([hidden, r2] . Wg + bg);  out = hidden + gate * r2
// One block per row; LDS tree reduction for the 2048-wide dot.
// ---------------------------------------------------------------------------
__global__ __launch_bounds__(256) void gate_residual(
    const float* __restrict__ hidden, const float* __restrict__ r2,
    const float* __restrict__ Wg, const float* __restrict__ bg,
    float* __restrict__ out)
{
    __shared__ float red[256];
    __shared__ float sgate;
    const int row = blockIdx.x;
    const int tid = threadIdx.x;
    const float* h = hidden + (size_t)row * DD;
    const float* r = r2 + (size_t)row * DD;
    float s = 0.0f;
    for (int c = tid; c < DD; c += 256) s += h[c] * Wg[c] + r[c] * Wg[DD + c];
    red[tid] = s;
    __syncthreads();
    for (int st = 128; st > 0; st >>= 1) {
        if (tid < st) red[tid] += red[tid + st];
        __syncthreads();
    }
    if (tid == 0) {
        float x = red[0] + bg[0];
        sgate = 1.0f / (1.0f + __expf(-x));
    }
    __syncthreads();
    float g = sgate;
    for (int c = tid; c < DD; c += 256)
        out[(size_t)row * DD + c] = h[c] + g * r[c];
}

// ---------------------------------------------------------------------------
extern "C" void kernel_launch(void* const* d_in, const int* in_sizes, int n_in,
                              void* d_out, int out_size, void* d_ws, size_t ws_size,
                              hipStream_t stream) {
    (void)in_sizes; (void)n_in; (void)out_size; (void)ws_size;
    const float* hidden    = (const float*)d_in[0];
    const float* knowledge = (const float*)d_in[1];
    const float* Wq        = (const float*)d_in[2];
    const float* bq        = (const float*)d_in[3];
    const float* Wo        = (const float*)d_in[4];
    const float* bo        = (const float*)d_in[5];
    const float* Wg        = (const float*)d_in[6];
    const float* bg        = (const float*)d_in[7];
    float* out             = (float*)d_out;

    // workspace carve-up (~134 MB total)
    char* p = (char*)d_ws;
    unsigned short* hid_h = (unsigned short*)p;  p += (size_t)MM  * DD * 2;  // 16 MB
    unsigned short* kno_h = (unsigned short*)p;  p += (size_t)NVV * DD * 2;  // 16 MB
    unsigned short* Wq_h  = (unsigned short*)p;  p += (size_t)DD  * DD * 2;  //  2 MB
    unsigned short* Wo_h  = (unsigned short*)p;  p += (size_t)DD  * DD * 2;  //  2 MB
    unsigned short* q_h   = (unsigned short*)p;  p += (size_t)MM  * DD * 2;  // 16 MB
    float*          topw  = (float*)p;           p += (size_t)MM * TOPK * 4; //  1 MB
    int*            topi  = (int*)p;             p += (size_t)MM * TOPK * 4; //  1 MB
    float*          retF  = (float*)p;           p += (size_t)MM * DD * 4;   // 32 MB
    unsigned short* retH  = (unsigned short*)p;  p += (size_t)MM * DD * 2;   // 16 MB
    float*          r2    = (float*)p;           p += (size_t)MM * DD * 4;   // 32 MB

    // 1) fp32 -> bf16 conversions
    cvt_f32_bf16<<<1024, 256, 0, stream>>>(hidden,    hid_h, MM * DD);
    cvt_f32_bf16<<<1024, 256, 0, stream>>>(knowledge, kno_h, NVV * DD);
    cvt_f32_bf16<<<256,  256, 0, stream>>>(Wq,        Wq_h,  DD * DD);
    cvt_f32_bf16<<<256,  256, 0, stream>>>(Wo,        Wo_h,  DD * DD);

    // 2) q = hidden * Wq + bq    (bf16 out, feeds logits GEMM)
    gemm_bf16_wmma<false><<<dim3(DD / 64, MM / 64), 256, 0, stream>>>(
        hid_h, Wq_h, bq, nullptr, q_h, MM, DD, DD);

    // 3) fused logits + top-32 + softmax   (32 rows per block)
    attn_topk_softmax<<<MM / 32, 256, 0, stream>>>(q_h, kno_h, topw, topi);

    // 4) weighted gather (fp32) + bf16 copy
    gather_weighted<<<MM, 256, 0, stream>>>(knowledge, topw, topi, retF, retH);

    // 5) r2 = retrieved * Wo + bo   (fp32 out)
    gemm_bf16_wmma<true><<<dim3(DD / 64, MM / 64), 256, 0, stream>>>(
        retH, Wo_h, bo, r2, nullptr, MM, DD, DD);

    // 6) gated residual
    gate_residual<<<MM, 256, 0, stream>>>(hidden, r2, Wg, bg, out);
}
